// TransformerPlanner_2018634629431
// MI455X (gfx1250) — matrix operand order repaired and checked by reading, not compiled, not run on hardware
//
#include <hip/hip_runtime.h>
#include <hip/hip_bf16.h>

// ---------------------------------------------------------------------------
// Fused transformer waypoint planner for MI455X (gfx1250, wave32, WMMA).
// One wave = one batch element. All GEMMs via v_wmma_f32_16x16x32_f16.
// v3: GEMMs accumulate ALL N tiles in registers, then store once -> WMMA
//     chains are back-to-back (waits cover only B loads, stores batched
//     after the last WMMA). Occupancy is LDS-limited (1 WG/WGP -> 2 waves
//     per SIMD32), so up to 512 VGPRs/wave are available for accumulators.
// ---------------------------------------------------------------------------

typedef _Float16 v16h __attribute__((ext_vector_type(16)));
typedef _Float16 h8   __attribute__((ext_vector_type(8)));
typedef float    v8f  __attribute__((ext_vector_type(8)));

#define NB   32768
#define NT2  20     // kv tokens (2*NT)
#define NWP  3      // waypoint queries
#define DM   64
#define NH   4
#define HD   16

// padded strides (halfs): K + 8  -> 16B-aligned rows, bank-conflict-free
#define S32  40     // K=32 matrices
#define S64  72     // K=64 matrices
#define S128 136    // K=128 matrices

// ---- LDS weight area (offsets in halfs, rows padded) ----
static constexpr int WT_IPW2 = 0;                    // [64][S32]   = 2560
static constexpr int WT_A1   = 2560;                 // 4 x [64][S64] = 4*4608
static constexpr int WT_A2   = 2560 + 4 * 4608;      // 20992
static constexpr int WT_FFN1 = 20992 + 4 * 4608;     // 39424: [128][S64] = 9216
static constexpr int WT_FFN2 = 39424 + 9216;         // 48640: [64][S128] = 8704
static constexpr int WT_OPW1 = 48640 + 8704;         // 57344: [32][S64]  = 2304
static constexpr int WT_HALFS = 57344 + 2304;        // 59648

// ---- LDS f32 param area (offsets in floats) ----
static constexpr int P_QEMB = 0;     // 192
static constexpr int P_IPW1 = 192;   // [2][32]
static constexpr int P_IPB1 = 256;
static constexpr int P_IPB2 = 288;
static constexpr int P_ABIAS = 352;  // + l*256 + {q,k,v,o}*64
static constexpr int P_FFNB1 = 864;
static constexpr int P_FFNB2 = 992;
static constexpr int P_LN   = 1056;  // + l*128 (g), +64 (b);  l = 0,1,2
static constexpr int P_OPB1 = 1440;
static constexpr int P_OPW2 = 1472;  // [32][2]
static constexpr int P_OPB2 = 1536;
static constexpr int P_FLOATS = 1552;

// ---- LDS layout (bytes) ----
static constexpr int SM_P       = WT_HALFS * 2;                  // 119296
static constexpr int SM_WAVE    = SM_P + P_FLOATS * 4;           // 125504
static constexpr int SM_WAVE_AL = ((SM_WAVE + 127) / 128) * 128; // 125568
// per-wave scratch (byte offsets within wave area)
static constexpr int WV_FEATS = 0;       // [32][S64] f16            4608
static constexpr int WV_K     = 4608;    // [32][S64] f16 (also FFN hidden [16][S128])
static constexpr int WV_V     = 9216;    // [32][S64] f16 (also head hidden [16][32])
static constexpr int WV_X     = 13824;   // [16][S64] f16 (also MLP hidden [32][S32])
static constexpr int WV_O32   = 16384;   // [16][64] f32 GEMM outputs
static constexpr int WV_XQ    = 20480;   // [3][64]  f32 residual stream
static constexpr int WV_S     = 21248;   // 240 f32 scores
static constexpr int WV_PB    = 22272;   // 240 f32 probs
static constexpr int WV_BYTES = 23296;
static constexpr int SM_TOTAL = SM_WAVE_AL + 8 * WV_BYTES;       // 311936 < 320KB

// Wave-local ordering: one wave's DS ops are processed in order by hardware;
// we only need to stop the compiler from reordering across the comm point.
__device__ __forceinline__ void wave_sync() {
  __builtin_amdgcn_fence(__ATOMIC_ACQ_REL, "wavefront");
  __builtin_amdgcn_wave_barrier();
}

// A operand: row-major f16 [*][astride] in LDS. Lane L -> row m0+L%16.
// A layout (matches B): element e <-> K = e + 16*(L>=16):
// one contiguous 16-half run.
__device__ __forceinline__ v16h ldsA(const _Float16* a, int astride, int m0,
                                     int k0, int lrow, int lhi) {
  const _Float16* r = a + (m0 + lrow) * astride + k0 + 16 * lhi;
  h8 c0 = *(const h8*)r;
  h8 c1 = *(const h8*)(r + 8);
  v16h o;
#pragma unroll
  for (int i = 0; i < 8; ++i) { o[i] = c0[i]; o[i + 8] = c1[i]; }
  return o;
}

// B operand: weights stored TRANSPOSED [N][bstride] f16 in LDS. Lane L ->
// column n0+L%16; element e <-> K = e + 16*(L>=16): one 16-half run.
__device__ __forceinline__ v16h ldsB(const _Float16* bt, int bstride, int n0,
                                     int k0, int lrow, int lhi) {
  const _Float16* r = bt + (n0 + lrow) * bstride + k0 + 16 * lhi;
  h8 c0 = *(const h8*)r;
  h8 c1 = *(const h8*)(r + 8);
  v16h o;
#pragma unroll
  for (int i = 0; i < 8; ++i) { o[i] = c0[i]; o[i + 8] = c1[i]; }
  return o;
}

// One 16-row M tile: D = A x BT + bias. A register-cached; ALL N-tile
// accumulators kept in registers, stores batched after the WMMA chain.
template <int KT, int NTILES>
__device__ __forceinline__ void gemm16_f16(const _Float16* A, int astride,
                                           int m0, const _Float16* BT,
                                           int bstride, const float* bias,
                                           _Float16* out, int ostride,
                                           bool relu, int lrow, int lhi) {
  v16h a[KT];
#pragma unroll
  for (int kt = 0; kt < KT; ++kt) a[kt] = ldsA(A, astride, m0, kt * 32, lrow, lhi);
  v8f acc[NTILES] = {};
#pragma unroll
  for (int nt = 0; nt < NTILES; ++nt)
#pragma unroll
    for (int kt = 0; kt < KT; ++kt)
      acc[nt] = __builtin_amdgcn_wmma_f32_16x16x32_f16(
          false, a[kt], false, ldsB(BT, bstride, nt * 16, kt * 32, lrow, lhi),
          (short)0, acc[nt], false, false);
#pragma unroll
  for (int nt = 0; nt < NTILES; ++nt) {
    float bv = bias[nt * 16 + lrow];
#pragma unroll
    for (int i = 0; i < 8; ++i) {
      float x = acc[nt][i] + bv;
      if (relu) x = fmaxf(x, 0.0f);
      out[(m0 + i + 8 * lhi) * ostride + nt * 16 + lrow] = (_Float16)x;
    }
  }
}

// Same, f32 output.
template <int KT, int NTILES>
__device__ __forceinline__ void gemm16_f32(const _Float16* A, int astride,
                                           int m0, const _Float16* BT,
                                           int bstride, const float* bias,
                                           float* out, int ostride,
                                           int lrow, int lhi) {
  v16h a[KT];
#pragma unroll
  for (int kt = 0; kt < KT; ++kt) a[kt] = ldsA(A, astride, m0, kt * 32, lrow, lhi);
  v8f acc[NTILES] = {};
#pragma unroll
  for (int nt = 0; nt < NTILES; ++nt)
#pragma unroll
    for (int kt = 0; kt < KT; ++kt)
      acc[nt] = __builtin_amdgcn_wmma_f32_16x16x32_f16(
          false, a[kt], false, ldsB(BT, bstride, nt * 16, kt * 32, lrow, lhi),
          (short)0, acc[nt], false, false);
#pragma unroll
  for (int nt = 0; nt < NTILES; ++nt) {
    float bv = bias[nt * 16 + lrow];
#pragma unroll
    for (int i = 0; i < 8; ++i)
      out[(m0 + i + 8 * lhi) * ostride + nt * 16 + lrow] = acc[nt][i] + bv;
  }
}

// Dual-output GEMM sharing the A registers (K and V projections): two
// independent WMMA chains interleave; all 8 accumulator tiles in registers.
template <int KT, int NTILES>
__device__ __forceinline__ void gemm16x2_f16(
    const _Float16* A, int astride, int m0, const _Float16* BT0,
    const float* b0, _Float16* out0, const _Float16* BT1, const float* b1,
    _Float16* out1, int bstride, int ostride, int lrow, int lhi) {
  v16h a[KT];
#pragma unroll
  for (int kt = 0; kt < KT; ++kt) a[kt] = ldsA(A, astride, m0, kt * 32, lrow, lhi);
  v8f acc0[NTILES] = {}, acc1[NTILES] = {};
#pragma unroll
  for (int nt = 0; nt < NTILES; ++nt)
#pragma unroll
    for (int kt = 0; kt < KT; ++kt) {
      acc0[nt] = __builtin_amdgcn_wmma_f32_16x16x32_f16(
          false, a[kt], false, ldsB(BT0, bstride, nt * 16, kt * 32, lrow, lhi),
          (short)0, acc0[nt], false, false);
      acc1[nt] = __builtin_amdgcn_wmma_f32_16x16x32_f16(
          false, a[kt], false, ldsB(BT1, bstride, nt * 16, kt * 32, lrow, lhi),
          (short)0, acc1[nt], false, false);
    }
#pragma unroll
  for (int nt = 0; nt < NTILES; ++nt) {
    float bv0 = b0[nt * 16 + lrow], bv1 = b1[nt * 16 + lrow];
#pragma unroll
    for (int i = 0; i < 8; ++i) {
      int ro = (m0 + i + 8 * lhi) * ostride + nt * 16 + lrow;
      out0[ro] = (_Float16)(acc0[nt][i] + bv0);
      out1[ro] = (_Float16)(acc1[nt][i] + bv1);
    }
  }
}

// Stage a global f32 weight [K][N] into LDS f16 transposed [N][K+8].
__device__ __forceinline__ void stageWT(_Float16* dst, const float* src,
                                        int K, int N, int tid) {
  for (int i = tid; i < K * N; i += 256) {
    int n = i / K, k = i - n * K;
    dst[n * (K + 8) + k] = (_Float16)src[k * N + n];
  }
}
__device__ __forceinline__ void stageF(float* dst, const float* src, int n,
                                       int tid) {
  for (int i = tid; i < n; i += 256) dst[i] = src[i];
}

// residual + LayerNorm over D=64 for the 3 query rows (lanes 0..2 serial).
__device__ __forceinline__ void resid_ln(float* xq, const float* add,
                                         _Float16* x16, const float* g,
                                         const float* bb, int lane) {
  if (lane < NWP) {
    int r = lane;
    float mu = 0.f;
    for (int c = 0; c < DM; ++c) mu += xq[r * DM + c] + add[r * DM + c];
    mu *= (1.0f / DM);
    float var = 0.f;
    for (int c = 0; c < DM; ++c) {
      float d = xq[r * DM + c] + add[r * DM + c] - mu;
      var += d * d;
    }
    var *= (1.0f / DM);
    float rs = rsqrtf(var + 1e-5f);
    for (int c = 0; c < DM; ++c) {
      float y = (xq[r * DM + c] + add[r * DM + c] - mu) * rs * g[c] + bb[c];
      xq[r * DM + c] = y;
      x16[r * S64 + c] = (_Float16)y;
    }
  }
  wave_sync();
}

__global__ __launch_bounds__(256, 1) void planner_kernel(
    const float* __restrict__ tl, const float* __restrict__ tr,
    const float* __restrict__ qemb, const float* __restrict__ ip_w1,
    const float* __restrict__ ip_b1, const float* __restrict__ ip_w2,
    const float* __restrict__ ip_b2,
    const float* a1_wq, const float* a1_wk, const float* a1_wv, const float* a1_wo,
    const float* a1_bq, const float* a1_bk, const float* a1_bv, const float* a1_bo,
    const float* a2_wq, const float* a2_wk, const float* a2_wv, const float* a2_wo,
    const float* a2_bq, const float* a2_bk, const float* a2_bv, const float* a2_bo,
    const float* ffn_w1, const float* ffn_b1, const float* ffn_w2, const float* ffn_b2,
    const float* n1_g, const float* n1_b, const float* n2_g, const float* n2_b,
    const float* n3_g, const float* n3_b,
    const float* op_w1, const float* op_b1, const float* op_w2, const float* op_b2,
    float* __restrict__ out) {
  extern __shared__ char smem[];
  _Float16* WT = (_Float16*)smem;
  float* P = (float*)(smem + SM_P);
  const int tid = threadIdx.x;
  const int wave = tid >> 5;
  const int lane = tid & 31;
  const int lrow = lane & 15;
  const int lhi = lane >> 4;

  // ---- cooperative weight staging (once per block) ----
  stageWT(WT + WT_IPW2, ip_w2, 32, 64, tid);
  stageWT(WT + WT_A1 + 0 * 4608, a1_wq, 64, 64, tid);
  stageWT(WT + WT_A1 + 1 * 4608, a1_wk, 64, 64, tid);
  stageWT(WT + WT_A1 + 2 * 4608, a1_wv, 64, 64, tid);
  stageWT(WT + WT_A1 + 3 * 4608, a1_wo, 64, 64, tid);
  stageWT(WT + WT_A2 + 0 * 4608, a2_wq, 64, 64, tid);
  stageWT(WT + WT_A2 + 1 * 4608, a2_wk, 64, 64, tid);
  stageWT(WT + WT_A2 + 2 * 4608, a2_wv, 64, 64, tid);
  stageWT(WT + WT_A2 + 3 * 4608, a2_wo, 64, 64, tid);
  stageWT(WT + WT_FFN1, ffn_w1, 64, 128, tid);
  stageWT(WT + WT_FFN2, ffn_w2, 128, 64, tid);
  stageWT(WT + WT_OPW1, op_w1, 64, 32, tid);
  stageF(P + P_QEMB, qemb, NWP * DM, tid);
  stageF(P + P_IPW1, ip_w1, 64, tid);
  stageF(P + P_IPB1, ip_b1, 32, tid);
  stageF(P + P_IPB2, ip_b2, 64, tid);
  stageF(P + P_ABIAS + 0, a1_bq, 64, tid);
  stageF(P + P_ABIAS + 64, a1_bk, 64, tid);
  stageF(P + P_ABIAS + 128, a1_bv, 64, tid);
  stageF(P + P_ABIAS + 192, a1_bo, 64, tid);
  stageF(P + P_ABIAS + 256, a2_bq, 64, tid);
  stageF(P + P_ABIAS + 320, a2_bk, 64, tid);
  stageF(P + P_ABIAS + 384, a2_bv, 64, tid);
  stageF(P + P_ABIAS + 448, a2_bo, 64, tid);
  stageF(P + P_FFNB1, ffn_b1, 128, tid);
  stageF(P + P_FFNB2, ffn_b2, 64, tid);
  stageF(P + P_LN + 0, n1_g, 64, tid);
  stageF(P + P_LN + 64, n1_b, 64, tid);
  stageF(P + P_LN + 128, n2_g, 64, tid);
  stageF(P + P_LN + 192, n2_b, 64, tid);
  stageF(P + P_LN + 256, n3_g, 64, tid);
  stageF(P + P_LN + 320, n3_b, 64, tid);
  stageF(P + P_OPB1, op_b1, 32, tid);
  stageF(P + P_OPW2, op_w2, 64, tid);
  stageF(P + P_OPB2, op_b2, 2, tid);
  __syncthreads();

  // ---- per-wave scratch ----
  char* wvp = smem + SM_WAVE_AL + wave * WV_BYTES;
  _Float16* feats16 = (_Float16*)(wvp + WV_FEATS);
  _Float16* k16 = (_Float16*)(wvp + WV_K);
  _Float16* v16 = (_Float16*)(wvp + WV_V);
  _Float16* x16 = (_Float16*)(wvp + WV_X);
  float* o32 = (float*)(wvp + WV_O32);
  float* xq = (float*)(wvp + WV_XQ);
  float* sbuf = (float*)(wvp + WV_S);
  float* pbuf = (float*)(wvp + WV_PB);

  const int b = blockIdx.x * 8 + wave;
  if (b >= NB) return;

  // ---- input MLP layer 1 (K=2, VALU): h = relu(pts @ ip_w1 + b1) ----
  if (lane < NT2) {
    const float* src = (lane < 10) ? (tl + ((size_t)b * 10 + lane) * 2)
                                   : (tr + ((size_t)b * 10 + (lane - 10)) * 2);
    float px = src[0], py = src[1];
    _Float16* hrow = x16 + lane * S32;  // h stored [32][S32] in x-buffer
    for (int j = 0; j < 32; ++j) {
      float v = fmaf(px, P[P_IPW1 + j], fmaf(py, P[P_IPW1 + 32 + j], P[P_IPB1 + j]));
      hrow[j] = (_Float16)fmaxf(v, 0.0f);
    }
  }
  wave_sync();

  // ---- feats = h @ ip_w2 + b2  (M=20->32, K=32, N=64) ----
  for (int m0 = 0; m0 < 32; m0 += 16)
    gemm16_f16<1, 4>(x16, S32, m0, WT + WT_IPW2, S32, P + P_IPB2, feats16,
                     S64, false, lrow, lhi);
  wave_sync();

  // ---- init residual stream with qemb ----
  for (int i = lane; i < NWP * DM; i += 32) {
    int r = i / DM, c = i - r * DM;
    float qv = P[P_QEMB + i];
    xq[i] = qv;
    x16[r * S64 + c] = (_Float16)qv;
  }
  wave_sync();

  // ---- 2 cross-attention layers ----
  for (int l = 0; l < 2; ++l) {
    const _Float16* wq = WT + (l ? WT_A2 : WT_A1);
    const _Float16* wk = wq + 4608;
    const _Float16* wv_ = wk + 4608;
    const _Float16* wo = wv_ + 4608;
    const float* bq = P + P_ABIAS + l * 256;

    // K,V projections, fused (shared A): [20,64]@[64,64] x2
    for (int m0 = 0; m0 < 32; m0 += 16)
      gemm16x2_f16<2, 4>(feats16, S64, m0, wk, bq + 64, k16, wv_, bq + 128,
                         v16, S64, S64, lrow, lhi);
    // Q projection -> o32 (f32)
    gemm16_f32<2, 4>(x16, S64, 0, wq, S64, bq, o32, 64, lrow, lhi);
    wave_sync();

    // scores: lane = token
    if (lane < NT2) {
      for (int h = 0; h < NH; ++h)
        for (int qi = 0; qi < NWP; ++qi) {
          float s = 0.f;
          for (int d = 0; d < HD; ++d)
            s = fmaf(o32[qi * DM + h * HD + d],
                     (float)k16[lane * S64 + h * HD + d], s);
          sbuf[(h * NWP + qi) * NT2 + lane] = s * 0.25f;  // 1/sqrt(16)
        }
    }
    wave_sync();
    // softmax: lane = (head, query)
    if (lane < NH * NWP) {
      const float* s = sbuf + lane * NT2;
      float m = -1e30f;
      for (int t = 0; t < NT2; ++t) m = fmaxf(m, s[t]);
      float e[NT2], den = 0.f;
      for (int t = 0; t < NT2; ++t) { e[t] = __expf(s[t] - m); den += e[t]; }
      float inv = 1.0f / den;
      for (int t = 0; t < NT2; ++t) pbuf[lane * NT2 + t] = e[t] * inv;
    }
    wave_sync();
    // attn @ V -> x16 rows 0..2 (f16), lane = output dim (x2 halves)
    for (int qi = 0; qi < NWP; ++qi)
      for (int hv = 0; hv < 2; ++hv) {
        int dim = hv * 32 + lane;
        const float* p = pbuf + ((dim >> 4) * NWP + qi) * NT2;
        float acc = 0.f;
        for (int t = 0; t < NT2; ++t)
          acc = fmaf(p[t], (float)v16[t * S64 + dim], acc);
        x16[qi * S64 + dim] = (_Float16)acc;
      }
    wave_sync();
    // out projection -> o32
    gemm16_f32<2, 4>(x16, S64, 0, wo, S64, bq + 192, o32, 64, lrow, lhi);
    wave_sync();
    // residual + LN(l)
    resid_ln(xq, o32, x16, P + P_LN + l * 128, P + P_LN + l * 128 + 64, lane);
  }

  // ---- FFN: relu(x2@W1+b1)@W2+b2, residual, LN3 ----
  gemm16_f16<2, 8>(x16, S64, 0, WT + WT_FFN1, S64, P + P_FFNB1, k16, S128,
                   true, lrow, lhi);
  wave_sync();
  gemm16_f32<4, 4>(k16, S128, 0, WT + WT_FFN2, S128, P + P_FFNB2, o32, 64,
                   lrow, lhi);
  wave_sync();
  resid_ln(xq, o32, x16, P + P_LN + 256, P + P_LN + 320, lane);

  // ---- output head: relu(x3@op_w1+b1) @ op_w2 + b2 ----
  gemm16_f16<2, 2>(x16, S64, 0, WT + WT_OPW1, S64, P + P_OPB1, v16, 32, true,
                   lrow, lhi);
  wave_sync();
  if (lane < NWP * 2) {
    int r = lane >> 1, c = lane & 1;
    float acc = P[P_OPB2 + c];
    for (int j = 0; j < 32; ++j)
      acc = fmaf((float)v16[r * 32 + j], P[P_OPW2 + j * 2 + c], acc);
    out[((size_t)b * NWP + r) * 2 + c] = acc;
  }
}

extern "C" void kernel_launch(void* const* d_in, const int* in_sizes, int n_in,
                              void* d_out, int out_size, void* d_ws,
                              size_t ws_size, hipStream_t stream) {
  (void)in_sizes; (void)n_in; (void)out_size; (void)d_ws; (void)ws_size;
  const float* p[37];
  for (int i = 0; i < 37; ++i) p[i] = (const float*)d_in[i];
  planner_kernel<<<NB / 8, 256, SM_TOTAL, stream>>>(
      p[0], p[1], p[2], p[3], p[4], p[5], p[6],
      p[7], p[8], p[9], p[10], p[11], p[12], p[13], p[14],
      p[15], p[16], p[17], p[18], p[19], p[20], p[21], p[22],
      p[23], p[24], p[25], p[26],
      p[27], p[28], p[29], p[30], p[31], p[32],
      p[33], p[34], p[35], p[36], (float*)d_out);
}
